// GenDot_27221502722366
// MI455X (gfx1250) — compile-verified, matrix operand order checked by reading, not run on hardware
//
#include <hip/hip_runtime.h>

// ---------------------------------------------------------------------------
// S[b,i,j] = X[b,i,:] @ M @ Y[b,j,:]
// Pre-pass:  Xh/Xl = bf16split(X); Yh/Yl = bf16split(Y); Mht/Mlt = bf16split(M^T)
// Stage 1:   XMh/XMl = bf16split( (Xh+Xl) @ (Mht+Mlt)^T )   [3-term split WMMA]
// Stage 2:   S = (XMh+XMl) @ (Yh+Yl)^T                      [3-term split WMMA]
// GEMM uses double-buffered LDS fed by GLOBAL_LOAD_ASYNC_TO_LDS_B128.
// ---------------------------------------------------------------------------

typedef __bf16 bf16;
typedef __attribute__((ext_vector_type(16))) __bf16 v16bf;
typedef __attribute__((ext_vector_type(8)))  __bf16 v8bf;
typedef __attribute__((ext_vector_type(4)))  __bf16 v4bf;
typedef __attribute__((ext_vector_type(8)))  float  v8f;
typedef __attribute__((ext_vector_type(4)))  float  f32x4;

static constexpr int BS   = 16;
static constexpr int DIM  = 1024;
static constexpr int MT   = 128;
static constexpr int NT   = 128;
static constexpr int KT   = 32;
static constexpr int NTHR = 256;

// ----------------------------- gfx1250 async path --------------------------
__device__ __forceinline__ void async_b128(unsigned lds_addr, const void* gaddr) {
    asm volatile("global_load_async_to_lds_b128 %0, %1, off"
                 :: "v"(lds_addr), "v"((unsigned long long)(size_t)gaddr)
                 : "memory");
}
__device__ __forceinline__ void wait_async0() {
    asm volatile("s_wait_asynccnt 0" ::: "memory");
}

// ----------------------------- pre-pass kernels ----------------------------
__global__ __launch_bounds__(256)
void split_f32_to_bf16hl(const float* __restrict__ in, bf16* __restrict__ hi,
                         bf16* __restrict__ lo, long n4) {
    long i = blockIdx.x * 256L + threadIdx.x;
    if (i >= n4) return;
    f32x4 v = ((const f32x4*)in)[i];
    v4bf h, l;
#pragma unroll
    for (int q = 0; q < 4; ++q) {
        h[q] = (bf16)v[q];
        l[q] = (bf16)(v[q] - (float)h[q]);
    }
    ((v4bf*)hi)[i] = h;
    ((v4bf*)lo)[i] = l;
}

// out[n][k] = split(in[k][n]),  DIM x DIM
__global__ __launch_bounds__(256)
void transpose_split_f32(const float* __restrict__ in, bf16* __restrict__ ht,
                         bf16* __restrict__ lt) {
    __shared__ float tile[32][33];
    const int n0 = blockIdx.x * 32, k0 = blockIdx.y * 32;
    const int c = threadIdx.x & 31, r0 = threadIdx.x >> 5;
#pragma unroll
    for (int rr = 0; rr < 4; ++rr) {
        const int r = r0 + rr * 8;
        tile[r][c] = in[(size_t)(k0 + r) * DIM + n0 + c];
    }
    __syncthreads();
#pragma unroll
    for (int rr = 0; rr < 4; ++rr) {
        const int r = r0 + rr * 8;
        const float v = tile[c][r];
        const bf16 h = (bf16)v;
        ht[(size_t)(n0 + r) * DIM + k0 + c] = h;
        lt[(size_t)(n0 + r) * DIM + k0 + c] = (bf16)(v - (float)h);
    }
}

// ----------------------------- WMMA fragment load --------------------------
// 16-bit A/B fragment (ISA 7.12.2): lane L (half=L>>4): elems 0..7 -> k=8*half+e,
// elems 8..15 -> k=16+8*half+(e-8). Two 16B contiguous LDS reads.
__device__ __forceinline__ v16bf load_frag(const bf16 (*T)[KT], int row, int half) {
    v8bf x0 = *(const v8bf*)&T[row][8 * half];
    v8bf x1 = *(const v8bf*)&T[row][16 + 8 * half];
    v16bf r;
#pragma unroll
    for (int i = 0; i < 8; ++i) { r[i] = x0[i]; r[i + 8] = x1[i]; }
    return r;
}

// ----------------------------- main GEMM -----------------------------------
// C[m][n] = sum_k (Ah+Al)[m][k] * (Bh+Bl)[n][k]   (B pre-transposed, [n][k])
// SPLIT_OUT: write bf16 hi/lo pair (stage 1) instead of fp32 (stage 2).
template <bool SPLIT_OUT>
__global__ __launch_bounds__(NTHR)
void gemm_bf16split(const bf16* __restrict__ Ah_g, const bf16* __restrict__ Al_g,
                    const bf16* __restrict__ Bh_g, const bf16* __restrict__ Bl_g,
                    float* __restrict__ Cf, bf16* __restrict__ Ch,
                    bf16* __restrict__ Cl,
                    long strideA, long strideB, long strideC) {
    const int b = blockIdx.z;
    const bf16* Ah_b = Ah_g + (size_t)b * (size_t)strideA;
    const bf16* Al_b = Al_g + (size_t)b * (size_t)strideA;
    const bf16* Bh_b = Bh_g + (size_t)b * (size_t)strideB;
    const bf16* Bl_b = Bl_g + (size_t)b * (size_t)strideB;

    const int m0 = blockIdx.y * MT;
    const int n0 = blockIdx.x * NT;

    // double-buffered tiles: [buf][hi/lo][row][k]  -> 2*2*8KB * (A+B) = 64 KB
    __shared__ __align__(16) bf16 sA[2][2][MT][KT];
    __shared__ __align__(16) bf16 sB[2][2][NT][KT];

    const int tid  = threadIdx.x;
    const int lane = tid & 31;
    const int wave = tid >> 5;
    const int wr   = wave & 3;     // 4 wave-rows * 32 rows
    const int wc   = wave >> 2;    // 2 wave-cols * 64 cols
    const int half = lane >> 4;
    const int l16  = lane & 15;

    // async staging map: thread -> one row, one 32B half-row (2 x b128)
    const int crow = tid >> 1;           // 0..127
    const int cofs = (tid & 1) * 16;     // bf16 element offset 0 / 16

    auto stage = [&](int buf, int k0) {
        const bf16* ga = Ah_b + (size_t)(m0 + crow) * DIM + k0 + cofs;
        async_b128((unsigned)(size_t)&sA[buf][0][crow][cofs],     ga);
        async_b128((unsigned)(size_t)&sA[buf][0][crow][cofs + 8], ga + 8);
        ga = Al_b + (size_t)(m0 + crow) * DIM + k0 + cofs;
        async_b128((unsigned)(size_t)&sA[buf][1][crow][cofs],     ga);
        async_b128((unsigned)(size_t)&sA[buf][1][crow][cofs + 8], ga + 8);
        ga = Bh_b + (size_t)(n0 + crow) * DIM + k0 + cofs;
        async_b128((unsigned)(size_t)&sB[buf][0][crow][cofs],     ga);
        async_b128((unsigned)(size_t)&sB[buf][0][crow][cofs + 8], ga + 8);
        ga = Bl_b + (size_t)(n0 + crow) * DIM + k0 + cofs;
        async_b128((unsigned)(size_t)&sB[buf][1][crow][cofs],     ga);
        async_b128((unsigned)(size_t)&sB[buf][1][crow][cofs + 8], ga + 8);
    };

    v8f acc[2][4];
#pragma unroll
    for (int i = 0; i < 2; ++i)
#pragma unroll
        for (int j = 0; j < 4; ++j)
#pragma unroll
            for (int q = 0; q < 8; ++q) acc[i][j][q] = 0.0f;

    constexpr int TSTEPS = DIM / KT;
    stage(0, 0);

    for (int t = 0; t < TSTEPS; ++t) {
        const int cur = t & 1;
        wait_async0();          // my copies of tile t have landed in LDS
        __syncthreads();        // everyone's copies have landed
        if (t + 1 < TSTEPS) stage(cur ^ 1, (t + 1) * KT);  // overlap with WMMA

        v16bf afh[2], afl[2], bfh[4], bfl[4];
#pragma unroll
        for (int tm = 0; tm < 2; ++tm) {
            const int row = wr * 32 + tm * 16 + l16;
            afh[tm] = load_frag(sA[cur][0], row, half);
            afl[tm] = load_frag(sA[cur][1], row, half);
        }
#pragma unroll
        for (int tn = 0; tn < 4; ++tn) {
            const int col = wc * 64 + tn * 16 + l16;
            bfh[tn] = load_frag(sB[cur][0], col, half);
            bfl[tn] = load_frag(sB[cur][1], col, half);
        }

#pragma unroll
        for (int tm = 0; tm < 2; ++tm)
#pragma unroll
            for (int tn = 0; tn < 4; ++tn) {
                acc[tm][tn] = __builtin_amdgcn_wmma_f32_16x16x32_bf16(
                    false, afh[tm], false, bfh[tn], (short)0, acc[tm][tn], false, false);
                acc[tm][tn] = __builtin_amdgcn_wmma_f32_16x16x32_bf16(
                    false, afh[tm], false, bfl[tn], (short)0, acc[tm][tn], false, false);
                acc[tm][tn] = __builtin_amdgcn_wmma_f32_16x16x32_bf16(
                    false, afl[tm], false, bfh[tn], (short)0, acc[tm][tn], false, false);
            }
    }

    // epilogue: C/D layout (VGPR v -> row v + 8*half, col = l16)
#pragma unroll
    for (int tm = 0; tm < 2; ++tm) {
        const int rbase = m0 + wr * 32 + tm * 16 + 8 * half;
#pragma unroll
        for (int tn = 0; tn < 4; ++tn) {
            const int col = n0 + wc * 64 + tn * 16 + l16;
#pragma unroll
            for (int v = 0; v < 8; ++v) {
                const size_t idx = (size_t)b * (size_t)strideC +
                                   (size_t)(rbase + v) * DIM + col;
                if (SPLIT_OUT) {
                    const float x = acc[tm][tn][v];
                    const bf16 h = (bf16)x;
                    Ch[idx] = h;
                    Cl[idx] = (bf16)(x - (float)h);
                } else {
                    Cf[idx] = acc[tm][tn][v];
                }
            }
        }
    }
}

// ----------------------------- host launcher -------------------------------
extern "C" void kernel_launch(void* const* d_in, const int* in_sizes, int n_in,
                              void* d_out, int out_size, void* d_ws, size_t ws_size,
                              hipStream_t stream) {
    (void)in_sizes; (void)n_in; (void)out_size; (void)ws_size;
    const float* X = (const float*)d_in[0];   // (BS,1024,1024)
    const float* Y = (const float*)d_in[1];   // (BS,1024,1024)
    const float* M = (const float*)d_in[2];   // (1024,1024)
    float* S = (float*)d_out;

    const size_t NXY = (size_t)BS * DIM * DIM;   // 16M elements
    const size_t NM  = (size_t)DIM * DIM;        // 1M elements

    // workspace layout (bf16 arrays), total = (6*NXY + 2*NM)*2 B = 196 MB
    char* p = (char*)d_ws;
    bf16* Xh  = (bf16*)p; p += NXY * sizeof(bf16);
    bf16* Xl  = (bf16*)p; p += NXY * sizeof(bf16);
    bf16* Yh  = (bf16*)p; p += NXY * sizeof(bf16);
    bf16* Yl  = (bf16*)p; p += NXY * sizeof(bf16);
    bf16* Mht = (bf16*)p; p += NM  * sizeof(bf16);
    bf16* Mlt = (bf16*)p; p += NM  * sizeof(bf16);
    bf16* XMh = (bf16*)p; p += NXY * sizeof(bf16);
    bf16* XMl = (bf16*)p; p += NXY * sizeof(bf16);

    // pre-pass: split X, Y; transpose+split M
    {
        const long n4 = (long)(NXY / 4);
        const int blocks = (int)(n4 / 256);
        split_f32_to_bf16hl<<<blocks, 256, 0, stream>>>(X, Xh, Xl, n4);
        split_f32_to_bf16hl<<<blocks, 256, 0, stream>>>(Y, Yh, Yl, n4);
        transpose_split_f32<<<dim3(32, 32), 256, 0, stream>>>(M, Mht, Mlt);
    }

    dim3 g(DIM / NT, DIM / MT, BS);
    dim3 blk(NTHR);
    // Stage 1: XM[b] = X[b] @ M   (strideB = 0: M^T shared, L2-resident)
    gemm_bf16split<true><<<g, blk, 0, stream>>>(
        Xh, Xl, Mht, Mlt, nullptr, XMh, XMl, (long)NM, 0L, (long)NM);
    // Stage 2: S[b] = XM[b] @ Y[b]^T
    gemm_bf16split<false><<<g, blk, 0, stream>>>(
        XMh, XMl, Yh, Yl, S, nullptr, nullptr, (long)NM, (long)NM, (long)NM);
}
